// HyperEdgeLayer_79096117723838
// MI455X (gfx1250) — matrix operand order repaired and checked by reading, not compiled, not run on hardware
//
#include <hip/hip_runtime.h>

typedef __attribute__((ext_vector_type(16))) _Float16 v16h;
typedef __attribute__((ext_vector_type(8)))  _Float16 v8h;
typedef __attribute__((ext_vector_type(8)))  float    v8f;

#define NN    40
#define FDIM  128
#define HDIM  128
#define EPB   (NN * NN * NN)   /* 64000 edges per batch  */
#define BB    4
#define ETOT  (BB * EPB)       /* 256000 total edges     */
#define TILE  128              /* edges per workgroup    */

// ---------------- helpers ----------------

// monotone float<->uint key for atomicMax-based float max
__device__ __forceinline__ unsigned fkey(float f) {
  int i = __float_as_int(f);
  return (i >= 0) ? ((unsigned)i | 0x80000000u) : ~((unsigned)i);
}
__device__ __forceinline__ float fdec(unsigned u) {
  int i = (u & 0x80000000u) ? (int)(u & 0x7fffffffu) : (int)(~u);
  return __int_as_float(i);
}

// A fragment (16x32 f16): lane L -> row M = mbase+(L&15); K = kb..kb+15, kb = 16*(L>=16)
__device__ __forceinline__ v16h load_A(const _Float16* sA, int mbase, int lane, int kbase, int ld) {
  return *(const v16h*)(sA + (mbase + (lane & 15)) * ld + kbase + ((lane >> 4) << 4));
}

// B fragment (32x16 f16) from n-major (transposed) weight tile: lane L -> col N = nbase+(L&15); K = 16-run
__device__ __forceinline__ v16h load_B(const _Float16* sB, int nbase, int lane, int kbase, int ld) {
  return *(const v16h*)(sB + (nbase + (lane & 15)) * ld + kbase + ((lane >> 4) << 4));
}

__device__ __forceinline__ v8f wmma_f16(v16h a, v16h b, v8f c) {
  return __builtin_amdgcn_wmma_f32_16x16x32_f16(false, a, false, b, (short)0, c, false, false);
}

__device__ __forceinline__ bool edge_valid(int ei, int b, const unsigned char* mask) {
  int i = ei / (NN * NN);
  int j = (ei / NN) % NN;
  int k = ei % NN;
  return (i < j) && (j < k) && mask[b * NN + i] && mask[b * NN + j] && mask[b * NN + k];
}

// ---------------- K0: convert + transpose weights to f16 (n-major: dst[n*K+k] = src[k*N+n]) ----------------
__global__ void k_convert_T(const float* __restrict__ src, _Float16* __restrict__ dst, int K, int Nn) {
  int idx = blockIdx.x * 256 + threadIdx.x;
  if (idx < K * Nn) {
    int n = idx / K;
    int k = idx % K;
    dst[idx] = (_Float16)src[k * Nn + n];
  }
}

// ---------------- init softmax stats ----------------
__global__ void k_init_stats(unsigned* __restrict__ maxbuf, float* __restrict__ sumbuf) {
  int t = blockIdx.x * 256 + threadIdx.x;
  if (t < BB * HDIM) { maxbuf[t] = 0u; sumbuf[t] = 0.0f; }
}

// ---------------- K1: xf = relu(X@Wi1+bi1)@Wi2+bi2  (128 edges per WG, 8 waves) ----------------
__global__ __launch_bounds__(256) void k_inner_mlp(
    const float* __restrict__ x, const float* __restrict__ bi1, const float* __restrict__ bi2,
    const _Float16* __restrict__ wi1T, const _Float16* __restrict__ wi2T,
    float* __restrict__ xf)
{
  __shared__ _Float16 sX[TILE * FDIM];      // 32 KB activation tile (f16)
  __shared__ _Float16 sT[TILE * HDIM];      // 32 KB relu intermediate
  __shared__ _Float16 sW[FDIM * HDIM];      // 32 KB weight buffer (Wi1 then Wi2)
  __shared__ float    sB1[HDIM], sB2[HDIM];

  const int tid = threadIdx.x, lane = tid & 31, wave = tid >> 5;
  const long ebase = (long)blockIdx.x * TILE;

  // stage Wi1^T and biases, convert x tile f32 -> f16
  {
    const uint4* wsrc = (const uint4*)wi1T;
    uint4* wdst = (uint4*)sW;
    for (int t = tid; t < 2048; t += 256) wdst[t] = wsrc[t];
    if (tid < HDIM) { sB1[tid] = bi1[tid]; sB2[tid] = bi2[tid]; }
    const float4* xsrc = (const float4*)(x + ebase * FDIM);
    for (int t = tid; t < 4096; t += 256) {
      float4 v = xsrc[t];
      int o = t * 4;
      sX[o + 0] = (_Float16)v.x; sX[o + 1] = (_Float16)v.y;
      sX[o + 2] = (_Float16)v.z; sX[o + 3] = (_Float16)v.w;
    }
  }
  __syncthreads();

  const int mbase = wave * 16;
  // GEMM1: sT = relu(sX @ Wi1 + bi1)
  for (int nt = 0; nt < 8; ++nt) {
    v8f acc = {};
#pragma unroll
    for (int kc = 0; kc < 4; ++kc) {
      v16h a = load_A(sX, mbase, lane, kc * 32, FDIM);
      v16h b = load_B(sW, nt * 16, lane, kc * 32, FDIM);
      acc = wmma_f16(a, b, acc);
    }
    int n  = nt * 16 + (lane & 15);
    int mo = mbase + ((lane >> 4) << 3);
    float bias = sB1[n];
#pragma unroll
    for (int r = 0; r < 8; ++r) {
      float v = acc[r] + bias;
      sT[(mo + r) * HDIM + n] = (_Float16)(v > 0.0f ? v : 0.0f);
    }
  }
  __syncthreads();
  // swap in Wi2^T
  {
    const uint4* wsrc = (const uint4*)wi2T;
    uint4* wdst = (uint4*)sW;
    for (int t = tid; t < 2048; t += 256) wdst[t] = wsrc[t];
  }
  __syncthreads();
  // GEMM2: xf = sT @ Wi2 + bi2
  for (int nt = 0; nt < 8; ++nt) {
    v8f acc = {};
#pragma unroll
    for (int kc = 0; kc < 4; ++kc) {
      v16h a = load_A(sT, mbase, lane, kc * 32, HDIM);
      v16h b = load_B(sW, nt * 16, lane, kc * 32, HDIM);
      acc = wmma_f16(a, b, acc);
    }
    int n  = nt * 16 + (lane & 15);
    int mo = mbase + ((lane >> 4) << 3);
    float bias = sB2[n];
#pragma unroll
    for (int r = 0; r < 8; ++r) {
      xf[(ebase + mo + r) * HDIM + n] = acc[r] + bias;
    }
  }
}

// ---------------- K2a/K2b: per-(batch,channel) masked softmax stats ----------------
__global__ __launch_bounds__(128) void k_softmax_max(
    const float* __restrict__ xf, const unsigned char* __restrict__ mask, unsigned* __restrict__ maxbuf)
{
  const int b = blockIdx.y, ch = threadIdx.x;
  const int ebeg = blockIdx.x * 256;
  float m = -3.402823e38f;
  for (int r = 0; r < 256; ++r) {
    int ei = ebeg + r;
    if (edge_valid(ei, b, mask)) {
      float v = xf[((long)b * EPB + ei) * HDIM + ch];
      m = fmaxf(m, v);
    }
  }
  atomicMax(&maxbuf[b * HDIM + ch], fkey(m));
}

__global__ __launch_bounds__(128) void k_softmax_sum(
    const float* __restrict__ xf, const unsigned char* __restrict__ mask,
    const unsigned* __restrict__ maxbuf, float* __restrict__ sumbuf)
{
  const int b = blockIdx.y, ch = threadIdx.x;
  const int ebeg = blockIdx.x * 256;
  const float mx = fdec(maxbuf[b * HDIM + ch]);
  float s = 0.0f;
  for (int r = 0; r < 256; ++r) {
    int ei = ebeg + r;
    if (edge_valid(ei, b, mask)) {
      float v = xf[((long)b * EPB + ei) * HDIM + ch];
      s += __expf(v - mx);
    }
  }
  atomicAdd(&sumbuf[b * HDIM + ch], s);
}

// ---------------- K3: out = relu([a*relu(xf@We), xf] @ Wo1 + bo1) @ Wo2 + bo2 ----------------
__global__ __launch_bounds__(256) void k_output(
    const float* __restrict__ xf, const unsigned char* __restrict__ mask,
    const unsigned* __restrict__ maxbuf, const float* __restrict__ sumbuf,
    const _Float16* __restrict__ weT, const _Float16* __restrict__ wo1T,
    const float* __restrict__ bo1, const float* __restrict__ wo2, const float* __restrict__ bo2,
    float* __restrict__ out)
{
  __shared__ _Float16 sXf[TILE * HDIM];        // 32 KB xf tile (f16)
  __shared__ _Float16 sXi[TILE * HDIM];        // 32 KB a * relu(xf@We)
  __shared__ _Float16 sW[64 * 256];            // 32 KB weight stage (We halves / Wo1 quarters)
  __shared__ float sMx[HDIM], sSum[HDIM], sBo1[2 * HDIM], sWo2[2 * HDIM];
  __shared__ unsigned char sValid[TILE];

  const int tid = threadIdx.x, lane = tid & 31, wave = tid >> 5;
  const long ebase = (long)blockIdx.x * TILE;
  const int b = (int)(ebase / EPB);            // 128 | 64000 -> tile never straddles batches

  if (tid < HDIM) {
    sMx[tid]  = fdec(maxbuf[b * HDIM + tid]);
    sSum[tid] = sumbuf[b * HDIM + tid];
  }
  if (tid < 2 * HDIM) { sBo1[tid] = bo1[tid]; sWo2[tid] = wo2[tid]; }
  if (tid < TILE) {
    int ei = (int)(ebase - (long)b * EPB) + tid;
    sValid[tid] = edge_valid(ei, b, mask) ? 1 : 0;
  }
  {
    const float4* src = (const float4*)(xf + ebase * HDIM);
    for (int t = tid; t < 4096; t += 256) {
      float4 v = src[t];
      int o = t * 4;
      sXf[o + 0] = (_Float16)v.x; sXf[o + 1] = (_Float16)v.y;
      sXf[o + 2] = (_Float16)v.z; sXf[o + 3] = (_Float16)v.w;
    }
  }
  __syncthreads();

  const int mbase = wave * 16;
  const int mo    = mbase + ((lane >> 4) << 3);

  // Phase 1: sXi = a * relu(sXf @ We), in two 64-column halves of We
  for (int h = 0; h < 2; ++h) {
    {
      const uint4* src = (const uint4*)(weT + h * 64 * HDIM);
      uint4* dst = (uint4*)sW;
      for (int t = tid; t < 1024; t += 256) dst[t] = src[t];
    }
    __syncthreads();
    for (int nt = 0; nt < 4; ++nt) {
      v8f acc = {};
#pragma unroll
      for (int kc = 0; kc < 4; ++kc) {
        v16h a = load_A(sXf, mbase, lane, kc * 32, HDIM);
        v16h w = load_B(sW, nt * 16, lane, kc * 32, HDIM);
        acc = wmma_f16(a, w, acc);
      }
      int n = h * 64 + nt * 16 + (lane & 15);
      float mx = sMx[n], sm = sSum[n];
#pragma unroll
      for (int r = 0; r < 8; ++r) {
        int m = mo + r;
        float xi = acc[r]; xi = xi > 0.0f ? xi : 0.0f;
        float av = 0.0f;
        if (sValid[m]) av = __expf((float)sXf[m * HDIM + n] - mx) / sm;
        sXi[m * HDIM + n] = (_Float16)(av * xi);
      }
    }
    __syncthreads();
  }

  // Phase 2: stream Wo1 in four 64-column quarters; fuse relu + *Wo2 + 16-lane reduce
  float rowacc[8];
#pragma unroll
  for (int r = 0; r < 8; ++r) rowacc[r] = 0.0f;

  for (int q = 0; q < 4; ++q) {
    {
      const uint4* src = (const uint4*)(wo1T + (long)q * 64 * 256);
      uint4* dst = (uint4*)sW;
      for (int t = tid; t < 2048; t += 256) dst[t] = src[t];
    }
    __syncthreads();
    for (int nt = 0; nt < 4; ++nt) {
      v8f acc = {};
#pragma unroll
      for (int kc = 0; kc < 4; ++kc) {        // K rows 0..127 of Wo1 multiply a*x_inner
        v16h a = load_A(sXi, mbase, lane, kc * 32, HDIM);
        v16h w = load_B(sW, nt * 16, lane, kc * 32, 256);
        acc = wmma_f16(a, w, acc);
      }
#pragma unroll
      for (int kc = 0; kc < 4; ++kc) {        // K rows 128..255 multiply xf
        v16h a = load_A(sXf, mbase, lane, kc * 32, HDIM);
        v16h w = load_B(sW, nt * 16, lane, kc * 32 + 128, 256);
        acc = wmma_f16(a, w, acc);
      }
      int n = q * 64 + nt * 16 + (lane & 15);
      float bias = sBo1[n], w2 = sWo2[n];
#pragma unroll
      for (int r = 0; r < 8; ++r) {
        float g = acc[r] + bias; g = g > 0.0f ? g : 0.0f;
        float c = g * w2;
#pragma unroll
        for (int off = 1; off < 16; off <<= 1) c += __shfl_xor(c, off, 32);
        rowacc[r] += c;                        // lanes 0-15: row mo+r ; lanes 16-31: row mo+r = mbase+8+r
      }
    }
    __syncthreads();
  }

  if ((lane & 15) == 0) {
    float b2 = bo2[0];
#pragma unroll
    for (int r = 0; r < 8; ++r) out[ebase + mo + r] = rowacc[r] + b2;
  }
}

// ---------------- host ----------------
extern "C" void kernel_launch(void* const* d_in, const int* in_sizes, int n_in,
                              void* d_out, int out_size, void* d_ws, size_t ws_size,
                              hipStream_t stream) {
  const float*         x    = (const float*)d_in[0];
  const unsigned char* mask = (const unsigned char*)d_in[1];
  const float* Wi1 = (const float*)d_in[2];
  const float* bi1 = (const float*)d_in[3];
  const float* Wi2 = (const float*)d_in[4];
  const float* bi2 = (const float*)d_in[5];
  const float* We  = (const float*)d_in[6];
  const float* Wo1 = (const float*)d_in[7];
  const float* bo1 = (const float*)d_in[8];
  const float* Wo2 = (const float*)d_in[9];
  const float* bo2 = (const float*)d_in[10];
  float* out = (float*)d_out;

  char* ws = (char*)d_ws;
  size_t off = 0;
  float* xf = (float*)(ws + off);            off += (size_t)ETOT * HDIM * sizeof(float); // 131 MB
  unsigned* maxbuf = (unsigned*)(ws + off);  off += BB * HDIM * sizeof(unsigned);
  float* sumbuf = (float*)(ws + off);        off += BB * HDIM * sizeof(float);
  _Float16* wi1T = (_Float16*)(ws + off);    off += FDIM * HDIM * sizeof(_Float16);
  _Float16* wi2T = (_Float16*)(ws + off);    off += FDIM * HDIM * sizeof(_Float16);
  _Float16* weT  = (_Float16*)(ws + off);    off += HDIM * HDIM * sizeof(_Float16);
  _Float16* wo1T = (_Float16*)(ws + off);    off += 2 * HDIM * 2 * HDIM * sizeof(_Float16);

  k_convert_T<<<(FDIM * FDIM + 255) / 256, 256, 0, stream>>>(Wi1, wi1T, FDIM, FDIM);
  k_convert_T<<<(FDIM * HDIM + 255) / 256, 256, 0, stream>>>(Wi2, wi2T, FDIM, HDIM);
  k_convert_T<<<(HDIM * HDIM + 255) / 256, 256, 0, stream>>>(We,  weT,  HDIM, HDIM);
  k_convert_T<<<(4 * HDIM * HDIM + 255) / 256, 256, 0, stream>>>(Wo1, wo1T, 2 * HDIM, 2 * HDIM);
  k_init_stats<<<2, 256, 0, stream>>>(maxbuf, sumbuf);

  k_inner_mlp<<<ETOT / TILE, 256, 0, stream>>>(x, bi1, bi2, wi1T, wi2T, xf);

  dim3 g2(EPB / 256, BB);
  k_softmax_max<<<g2, 128, 0, stream>>>(xf, mask, maxbuf);
  k_softmax_sum<<<g2, 128, 0, stream>>>(xf, mask, maxbuf, sumbuf);

  k_output<<<ETOT / TILE, 256, 0, stream>>>(xf, mask, maxbuf, sumbuf, weT, wo1T,
                                            bo1, Wo2, bo2, out);
}
